// AttentiveStatsPooling_72859825209727
// MI455X (gfx1250) — matrix operand order repaired and checked by reading, not compiled, not run on hardware
//
#include <hip/hip_runtime.h>
#include <math.h>

// Attentive stats pooling, fused for MI455X (gfx1250, wave32, WMMA bf16).
// [convert W -> bf16] then, per group of 8 batches (x stays L2-resident):
//   gemm1_tanh:  e(B,T,BN) bf16 = tanh(W_tdnn @ x + b_tdnn), stored transposed
//   gemm2_stats: a = W_attn @ e fused with masked softmax + weighted mean/std
// b_attn dropped (softmax shift-invariant per channel); no online max needed
// since |a| <= ~sum|W_attn row| ~ 9, so exp(a) cannot overflow f32.

typedef __bf16        v16bf __attribute__((ext_vector_type(16)));
typedef float         v8f   __attribute__((ext_vector_type(8)));
typedef unsigned int  ui4   __attribute__((ext_vector_type(4)));

#define B_    32
#define C_    1536
#define T_    2048
#define BN_   128
#define BGRP_ 8          // batches per launch group (8*12.6MB = 101MB < 192MB L2)

__device__ __forceinline__ unsigned short f2bf(float f) {
  union { float f; unsigned u; } v; v.f = f;
  unsigned r = v.u + 0x7FFFu + ((v.u >> 16) & 1u);   // round-to-nearest-even
  return (unsigned short)(r >> 16);
}

__device__ __forceinline__ v16bf mk_frag(ui4 lo, ui4 hi) {
  struct P { ui4 a, b; } p; p.a = lo; p.b = hi;
  return __builtin_bit_cast(v16bf, p);
}

__device__ __forceinline__ float fast_tanh(float x) {
#if __has_builtin(__builtin_amdgcn_tanhf)
  return __builtin_amdgcn_tanhf(x);                  // native v_tanh_f32
#elif __has_builtin(__builtin_amdgcn_tanh_f32)
  return __builtin_amdgcn_tanh_f32(x);
#else
  // branch-free: tanh(|x|) = (1-e)/(1+e), e = exp(-2|x|); restore sign
  float e = __expf(-2.0f * fabsf(x));
  float t = (1.0f - e) * __builtin_amdgcn_rcpf(1.0f + e);
  return copysignf(t, x);
#endif
}

// ---------------- Kernel 0: f32 -> bf16 weight conversion --------------------
__global__ void __launch_bounds__(256)
convert_weights(const float* __restrict__ Wt, const float* __restrict__ Wa,
                unsigned short* __restrict__ wt_bf, unsigned short* __restrict__ wa_bf) {
  int i = blockIdx.x * blockDim.x + threadIdx.x;
  if (i < BN_ * C_) {
    wt_bf[i] = f2bf(Wt[i]);
    wa_bf[i] = f2bf(Wa[i]);
  }
}

// ---------------- Kernel 1: e = tanh(W_tdnn @ x + b_tdnn) --------------------
// Grid: (T/64, BGRP_). 8 waves; wave w owns channels 16w..16w+15 x 64 t.
// e stored TRANSPOSED as (B, T, BN) bf16 so kernel 2's B-fragments are
// 16B global loads (K-contiguous).
__global__ void __launch_bounds__(256)
gemm1_tanh(const float* __restrict__ x, const unsigned short* __restrict__ wt,
           const float* __restrict__ btdnn, unsigned short* __restrict__ e_t,
           int b0) {
  const int tblk = blockIdx.x;
  const int b    = b0 + blockIdx.y;
  const int tid  = threadIdx.x;
  const int wave = tid >> 5;
  const int lane = tid & 31;
  const int m    = lane & 15;    // A row / B column / D column
  const int half = lane >> 4;    // A khalf / B k-group / D row-half
  const int t0   = tblk * 64;
  const int ch0  = wave * 16;

  // x tile transposed [t][c]; row stride 56 halves = 112B (16B-aligned,
  // conflict-free: n*28 mod 64 distinct with spacing 4 for n=0..15)
  __shared__ __align__(16) unsigned short xs[64 * 56];

  v8f acc[4];
#pragma unroll
  for (int i = 0; i < 4; i++) acc[i] = (v8f){0,0,0,0,0,0,0,0};

  const int cload = tid >> 3;        // 0..31: channel within K-step
  const int t8    = (tid & 7) * 8;   // 8 consecutive t per thread

  for (int kb = 0; kb < C_; kb += 32) {
    // stage x (32c x 64t) -> LDS transposed, converting to bf16
    const float* xp = x + ((size_t)b * C_ + kb + cload) * T_ + t0 + t8;
    float xv[8];
#pragma unroll
    for (int j = 0; j < 8; j++) xv[j] = xp[j];
#pragma unroll
    for (int j = 0; j < 8; j++) xs[(t8 + j) * 56 + cload] = f2bf(xv[j]);
    __syncthreads();

    // A fragment: W_tdnn row ch0+m; K = kb + half*8 + {0..7} and +16
    const unsigned short* wp = wt + (size_t)(ch0 + m) * C_ + kb + half * 8;
    v16bf afrag = mk_frag(*(const ui4*)wp, *(const ui4*)(wp + 16));

#pragma unroll
    for (int nt = 0; nt < 4; nt++) {
      const unsigned short* bp = &xs[(nt * 16 + m) * 56 + half * 8];
      v16bf bfrag = mk_frag(*(const ui4*)bp, *(const ui4*)(bp + 16));
      acc[nt] = __builtin_amdgcn_wmma_f32_16x16x32_bf16(
          false, afrag, false, bfrag, (short)0, acc[nt], false, false);
    }
    __syncthreads();
  }

  // D layout: lane holds column t=(lane&15), rows chbase..chbase+7
  const int chbase = ch0 + half * 8;
  float bias[8];
#pragma unroll
  for (int r = 0; r < 8; r++) bias[r] = btdnn[chbase + r];

#pragma unroll
  for (int nt = 0; nt < 4; nt++) {
    int t = t0 + nt * 16 + m;
    unsigned pk[4];
#pragma unroll
    for (int j = 0; j < 4; j++) {
      unsigned short l0 = f2bf(fast_tanh(acc[nt][2 * j]     + bias[2 * j]));
      unsigned short h0 = f2bf(fast_tanh(acc[nt][2 * j + 1] + bias[2 * j + 1]));
      pk[j] = (unsigned)l0 | ((unsigned)h0 << 16);
    }
    ui4 q = {pk[0], pk[1], pk[2], pk[3]};
    *(ui4*)(e_t + ((size_t)b * T_ + t) * BN_ + chbase) = q;   // 16B store
  }
}

// ------- Kernel 2: a = W_attn @ e, masked softmax, weighted mean/std --------
// Grid: (C/16, BGRP_). 8 waves; wave i handles t-chunks i, i+8, ... keeping
// plain (S0,S1,S2) sums per lane for its 8 channels. 'a' never hits memory.
__global__ void __launch_bounds__(256)
gemm2_softmax_stats(const float* __restrict__ x,
                    const unsigned char* __restrict__ mask,   // jnp bool = 1 byte
                    const unsigned short* __restrict__ wa,
                    const unsigned short* __restrict__ e_t,
                    float* __restrict__ out, int b0) {
  const int cblk = blockIdx.x;
  const int b    = b0 + blockIdx.y;
  const int tid  = threadIdx.x;
  const int wave = tid >> 5;
  const int lane = tid & 31;
  const int n    = lane & 15;
  const int g    = lane >> 4;
  const int c0   = cblk * 16;

  __shared__ float4 states[16][128];   // 32 KB partial sums per channel

  // A fragments (loop-invariant): W_attn rows c0..c0+15, K = 0..127
  v16bf afrag[4];
  {
    const unsigned short* wp = wa + (size_t)(c0 + n) * BN_ + g * 8;
#pragma unroll
    for (int f = 0; f < 4; f++)
      afrag[f] = mk_frag(*(const ui4*)(wp + f * 32),
                         *(const ui4*)(wp + f * 32 + 16));
  }

  const float* xrow = x + ((size_t)b * C_ + c0 + g * 8) * T_;

  float S0[8], S1[8], S2[8];
#pragma unroll
  for (int r = 0; r < 8; r++) { S0[r] = 0.f; S1[r] = 0.f; S2[r] = 0.f; }

  for (int chunk = wave; chunk < T_ / 16; chunk += 8) {
    const int t = chunk * 16 + n;
    // B fragments: e_t(B,T,BN) -> lane(n,g) reads K = f*32 + g*8 + {0..7} and +16
    const unsigned short* ep = e_t + ((size_t)b * T_ + t) * BN_ + g * 8;
    v8f acc = (v8f){0,0,0,0,0,0,0,0};
#pragma unroll
    for (int f = 0; f < 4; f++) {
      v16bf bfrag = mk_frag(*(const ui4*)(ep + f * 32),
                            *(const ui4*)(ep + f * 32 + 16));
      acc = __builtin_amdgcn_wmma_f32_16x16x32_bf16(
          false, afrag[f], false, bfrag, (short)0, acc, false, false);
    }
    // last-use reads: NT so they don't evict group-resident x
    const bool masked = __builtin_nontemporal_load(&mask[(size_t)b * T_ + t]) != 0;
#pragma unroll
    for (int r = 0; r < 8; r++) {
      // |a| <= ~9 so exp can't overflow; masked -> -60 (weight ~9e-27, still
      // yields exact uniform softmax if ALL positions are masked)
      float a  = masked ? -60.0f : acc[r];
      float xv = __builtin_nontemporal_load(&xrow[(size_t)r * T_ + t]);
      float p  = __expf(a);
      S0[r] += p;
      S1[r] += p * xv;
      S2[r] += p * xv * xv;
    }
  }

#pragma unroll
  for (int r = 0; r < 8; r++)
    states[r + 8 * g][wave * 16 + n] = make_float4(S0[r], S1[r], S2[r], 0.f);
  __syncthreads();

  if (tid < 16) {
    float s0 = 0.f, s1 = 0.f, s2 = 0.f;
    for (int s = 0; s < 128; s++) {
      float4 st = states[tid][s];
      s0 += st.x; s1 += st.y; s2 += st.z;
    }
    float mean = s1 / s0;
    float var  = s2 / s0 - mean * mean;
    float sd   = sqrtf(fmaxf(var, 1e-9f));
    out[(size_t)b * (2 * C_) + c0 + tid]      = mean;
    out[(size_t)b * (2 * C_) + C_ + c0 + tid] = sd;
  }
}

// ---------------------------------------------------------------------------
extern "C" void kernel_launch(void* const* d_in, const int* in_sizes, int n_in,
                              void* d_out, int out_size, void* d_ws, size_t ws_size,
                              hipStream_t stream) {
  const float*         x    = (const float*)d_in[0];
  const unsigned char* mask = (const unsigned char*)d_in[1];  // bool array
  const float*         Wt   = (const float*)d_in[2];
  const float*         btd  = (const float*)d_in[3];
  const float*         Wa   = (const float*)d_in[4];
  // d_in[5] (b_attn) intentionally unused: softmax is shift-invariant per channel.
  float* out = (float*)d_out;

  // Workspace: W_tdnn bf16 (384KB) | W_attn bf16 (384KB) | e_t (B,T,BN) bf16 (16.8MB)
  unsigned short* wt_bf = (unsigned short*)d_ws;
  unsigned short* wa_bf = wt_bf + (size_t)BN_ * C_;
  unsigned short* e_t   = wa_bf + (size_t)C_ * BN_;

  convert_weights<<<(BN_ * C_ + 255) / 256, 256, 0, stream>>>(Wt, Wa, wt_bf, wa_bf);

  // Interleave per batch-group so each group's x slab (101MB) stays in the
  // 192MB L2 between the two GEMM passes -> x hits HBM only once overall.
  for (int b0 = 0; b0 < B_; b0 += BGRP_) {
    gemm1_tanh<<<dim3(T_ / 64, BGRP_), 256, 0, stream>>>(x, wt_bf, btd, e_t, b0);
    gemm2_softmax_stats<<<dim3(C_ / 16, BGRP_), 256, 0, stream>>>(
        x, mask, wa_bf, e_t, out, b0);
  }
}
